// PaiNNAtomEncoder_56349970923945
// MI455X (gfx1250) — matrix operand order repaired and checked by reading, not compiled, not run on hardware
//
#include <hip/hip_runtime.h>
#include <cstdint>
#include <cstddef>

// ---------------- problem constants (from reference) ----------------
#define NB    4
#define NPER  4096
#define EPERB 49152
#define NTOT  (NB*NPER)        // 16384 atoms
#define ETOT  (NB*EPERB)       // 196608 edges
#define FDIM  128
#define KRBF  20
#define NLAY  3
#define CUTF  5.0f
#define PI_F  3.14159265358979323846f

// ---------------- WMMA types ----------------
typedef __attribute__((ext_vector_type(16))) __bf16 bf16x16;
typedef __attribute__((ext_vector_type(8)))  float  v8f;

union FragB16 { bf16x16 v; uint4 q[2]; };

__device__ __forceinline__ unsigned short f2bf(float f) {
  unsigned int u = __float_as_uint(f);
  unsigned int r = 0x7FFFu + ((u >> 16) & 1u);
  return (unsigned short)((u + r) >> 16);
}

__device__ __forceinline__ void atomAddF(float* p, float x) {
  __hip_atomic_fetch_add(p, x, __ATOMIC_RELAXED, __HIP_MEMORY_SCOPE_AGENT);
}

// ---- CDNA5 async global -> LDS copy (ASYNCcnt-tracked DMA path) ----
__device__ __forceinline__ void async_g2l_b128(const void* lds, const void* g) {
  uint32_t l = (uint32_t)(uintptr_t)lds;   // low 32 bits = LDS offset
  asm volatile("global_load_async_to_lds_b128 %0, %1, off"
               :: "v"(l), "v"(g) : "memory");
}
__device__ __forceinline__ void wait_async0() {
  asm volatile("s_wait_asynccnt 0x0" ::: "memory");
}

// ====================================================================
// GEMM: out[m,n] = act( sum_k A[m,k]*W[n,k] + bias[n] )   (X @ W^T)
// A: M x Kd bf16 row-major, W: Nd x Kd bf16 row-major.
// Block = 8 waves, all on the SAME 64-wide N tile; W tile (64 x Kd)
// async-DMA'd to LDS once (row stride padded +8 elems -> conflict-free
// ds_load_b128). Each wave computes 32(M) x 64(N): 8 WMMAs per K-step
// with A fragments reused across the 4 N subtiles.
// Requires: M % 256 == 0, Nd % 64 == 0, Kd % 32 == 0, Kd <= 256.
// ====================================================================
__global__ __launch_bounds__(256)
void painn_gemm_bf16(const unsigned short* __restrict__ A,
                     const unsigned short* __restrict__ W,
                     const float* __restrict__ bias,
                     float* __restrict__ outF,
                     unsigned short* __restrict__ outB,
                     int M, int Nd, int Kd, int act)
{
  __shared__ alignas(16) unsigned short sW[(256 + 8) * 64];

  const int t    = threadIdx.x;
  const int lane = t & 31;
  const int wv   = t >> 5;
  const int gn   = Nd >> 6;
  const int bn   = blockIdx.x % gn;
  const int bm   = blockIdx.x / gn;
  const int ldw  = Kd + 8;               // LDS row stride (elements)

  // ---- async-stage W tile: rows [bn*64, bn*64+64) x Kd ----
  {
    const int cpr   = Kd >> 3;           // 16B chunks per row
    const int total = 64 * cpr;
    const unsigned short* Wb = W + (size_t)(bn * 64) * Kd;
    for (int i = t; i < total; i += 256) {
      const int r = i / cpr, c = i - r * cpr;
      async_g2l_b128(&sW[r * ldw + c * 8], Wb + (size_t)r * Kd + c * 8);
    }
    wait_async0();
  }
  __syncthreads();

  const int hi  = lane >> 4;
  const int l16 = lane & 15;
  const int row0 = bm * 256 + wv * 32 + l16;
  const unsigned short* Arow0 = A + (size_t)row0 * Kd;
  const unsigned short* Arow1 = Arow0 + (size_t)16 * Kd;

  v8f zero = {0.f,0.f,0.f,0.f,0.f,0.f,0.f,0.f};
  v8f acc[2][4] = {{zero,zero,zero,zero},{zero,zero,zero,zero}};

  for (int k0 = 0; k0 < Kd; k0 += 32) {
    FragB16 a0, a1;
    const int ka = k0 + hi * 8;          // A: lanes 0-15 hold K+0..7 & K+16..23
    a0.q[0] = *(const uint4*)(Arow0 + ka);
    a0.q[1] = *(const uint4*)(Arow0 + ka + 16);
    a1.q[0] = *(const uint4*)(Arow1 + ka);
    a1.q[1] = *(const uint4*)(Arow1 + ka + 16);
    const int kb = k0 + hi * 8;          // B: lanes 0-15 hold K+0..7 & K+16..23
#pragma unroll
    for (int j = 0; j < 4; ++j) {
      const unsigned short* ls = &sW[(j * 16 + l16) * ldw + kb];
      FragB16 b;
      b.q[0] = *(const uint4*)(ls);
      b.q[1] = *(const uint4*)(ls + 16);
      acc[0][j] = __builtin_amdgcn_wmma_f32_16x16x32_bf16(
          false, a0.v, false, b.v, (short)0, acc[0][j], false, false);
      acc[1][j] = __builtin_amdgcn_wmma_f32_16x16x32_bf16(
          false, a1.v, false, b.v, (short)0, acc[1][j], false, false);
    }
  }

  // C layout: VGPR i -> M=i (lanes 0-15) / M=8+i (lanes 16-31), N = lane&15
#pragma unroll
  for (int p = 0; p < 2; ++p) {
    const int rbase = bm * 256 + wv * 32 + p * 16 + hi * 8;
#pragma unroll
    for (int j = 0; j < 4; ++j) {
      const int col = bn * 64 + j * 16 + l16;
      const float bval = bias ? bias[col] : 0.f;
#pragma unroll
      for (int i = 0; i < 8; ++i) {
        float x = acc[p][j][i] + bval;
        if (act) x = x / (1.f + __expf(-x));   // silu
        const size_t off = (size_t)(rbase + i) * Nd + col;
        if (outF) outF[off] = x;
        if (outB) outB[off] = f2bf(x);
      }
    }
  }
}

// ====================================================================
// fp32 -> bf16 convert
// ====================================================================
__global__ void painn_cvt_bf16(const float* __restrict__ src,
                               unsigned short* __restrict__ dst, int n)
{
  int i = blockIdx.x * blockDim.x + threadIdx.x;
  if (i < n) dst[i] = f2bf(src[i]);
}

// ====================================================================
// Edge geometry: dist, rbf (sinc basis), cosine cutoff, direction, indices
// ====================================================================
__global__ void painn_edge_geom(const float* __restrict__ xyz,
                                const float* __restrict__ cell,
                                const int*   __restrict__ aedges,
                                const float* __restrict__ edisp,
                                float* __restrict__ rbf,
                                float* __restrict__ cutv,
                                float* __restrict__ dirn,
                                int* __restrict__ idxS,
                                int* __restrict__ idxR)
{
  int e = blockIdx.x * blockDim.x + threadIdx.x;
  if (e >= ETOT) return;
  const int b = e / EPERB;
  const int s = aedges[2*e]     + b * NPER;
  const int r = aedges[2*e + 1] + b * NPER;
  const float* C = cell + b * 9;
  const float d0 = edisp[3*e], d1 = edisp[3*e+1], d2 = edisp[3*e+2];
  float diff[3];
  float dist2 = 0.f;
#pragma unroll
  for (int d = 0; d < 3; ++d) {
    const float disp = d0 * C[d] + d1 * C[3 + d] + d2 * C[6 + d];
    const float df = xyz[3*r + d] - (xyz[3*s + d] + disp);
    diff[d] = df;
    dist2 += df * df;
  }
  const float dist = sqrtf(dist2);
  idxS[e] = s; idxR[e] = r;
  cutv[e] = (dist < CUTF) ? 0.5f * (cosf(PI_F * dist / CUTF) + 1.f) : 0.f;
  const float inv = 1.f / fmaxf(dist, 1e-12f);
#pragma unroll
  for (int d = 0; d < 3; ++d) dirn[3*e + d] = diff[d] * inv;
  const float invd = 1.f / dist;
#pragma unroll
  for (int k = 0; k < KRBF; ++k)
    rbf[(size_t)e * KRBF + k] = sinf(dist * (float)(k + 1) * (PI_F / CUTF)) * invd;
}

// ====================================================================
// Build init MLP input h (bf16, N x 256): [emb(node) | V*pot_W + pot_b]
// ====================================================================
__global__ void painn_h_build(const int* __restrict__ nodes,
                              const float* __restrict__ emb,
                              const float* __restrict__ Vflat,
                              const float* __restrict__ potW,
                              const float* __restrict__ potB,
                              unsigned short* __restrict__ hbf)
{
  int t = blockIdx.x * blockDim.x + threadIdx.x;
  if (t >= NTOT * FDIM) return;
  const int n = t >> 7, f = t & 127;
  hbf[(size_t)n * 256 + f]       = f2bf(emb[(size_t)nodes[n] * FDIM + f]);
  hbf[(size_t)n * 256 + 128 + f] = f2bf(Vflat[n] * potW[f] + potB[f]);
}

// v[n,d,f] = E_flat[n,d] * vec_w[f]
__global__ void painn_v_init(const float* __restrict__ Eflat,
                             const float* __restrict__ vecw,
                             float* __restrict__ v)
{
  int t = blockIdx.x * blockDim.x + threadIdx.x;
  if (t >= 3 * NTOT * FDIM) return;
  const int f = t & 127;
  const int d = (t >> 7) % 3;
  const int n = t / 384;
  v[t] = Eflat[n * 3 + d] * vecw[f];
}

// ====================================================================
// Fused edge message + scatter. Filter GEMM (K=20) runs as VALU FMAs
// reading filt_W from LDS in TRANSPOSED padded layout sWT[k*392+f]
// (stride-1 across lanes -> conflict-free; k is a broadcast).
// Per-edge rbf/dir/cut/idx blocks staged via CDNA5 async global->LDS.
// Block = 32 edges x 128 features.
// ====================================================================
#define EPB 32
#define WTS 392          // padded transposed stride (384 + 8)
__global__ __launch_bounds__(256)
void painn_edge_msg(const float* __restrict__ rbf,
                    const float* __restrict__ cutv,
                    const float* __restrict__ dirn,
                    const int*   __restrict__ idxS,
                    const int*   __restrict__ idxR,
                    const float* __restrict__ so,
                    const float* __restrict__ s,
                    const float* __restrict__ v,
                    const float* __restrict__ fW,   // 384 x 20
                    const float* __restrict__ fb,   // 384
                    float* __restrict__ sacc,
                    float* __restrict__ vacc)
{
  __shared__ alignas(16) float sWT[KRBF * WTS];
  __shared__ float sB[384];
  __shared__ alignas(16) float sRbf[EPB * KRBF];
  __shared__ alignas(16) float sCut[EPB];
  __shared__ alignas(16) float sDir[EPB * 3];
  __shared__ alignas(16) int   sS[EPB];
  __shared__ alignas(16) int   sR[EPB];

  const int t  = threadIdx.x;
  const int e0 = blockIdx.x * EPB;

  // async-stage the contiguous per-edge blocks (16B chunks)
  if (t < 160) {                                   // rbf: 2560 B
    async_g2l_b128(&sRbf[t * 4], rbf + (size_t)e0 * KRBF + t * 4);
  } else if (t < 184) {                            // dirn: 384 B
    const int i = t - 160;
    async_g2l_b128(&sDir[i * 4], dirn + (size_t)e0 * 3 + i * 4);
  } else if (t < 192) {                            // cut: 128 B
    const int i = t - 184;
    async_g2l_b128(&sCut[i * 4], cutv + e0 + i * 4);
  } else if (t < 200) {                            // send idx: 128 B
    const int i = t - 192;
    async_g2l_b128(&sS[i * 4], idxS + e0 + i * 4);
  } else if (t < 208) {                            // recv idx: 128 B
    const int i = t - 200;
    async_g2l_b128(&sR[i * 4], idxR + e0 + i * 4);
  }
  // transpose-stage filt_W (coalesced global reads)
  for (int i = t; i < 384 * KRBF; i += 256) {
    const int f = i / KRBF, k = i - f * KRBF;
    sWT[k * WTS + f] = fW[i];
  }
  for (int i = t; i < 384; i += 256) sB[i] = fb[i];
  wait_async0();
  __syncthreads();

#pragma unroll
  for (int it = 0; it < (EPB * FDIM) / 256; ++it) {
    const int lin = it * 256 + t;
    const int el = lin >> 7;
    const int f  = lin & 127;
    const int si = sS[el], ri = sR[el];
    const float c = sCut[el];

    float fw0 = sB[f], fw1 = sB[128 + f], fw2 = sB[256 + f];
    const float* rb = sRbf + el * KRBF;
#pragma unroll
    for (int k = 0; k < KRBF; ++k) {
      const float rv = rb[k];
      fw0 = fmaf(rv, sWT[k * WTS + f],       fw0);
      fw1 = fmaf(rv, sWT[k * WTS + 128 + f], fw1);
      fw2 = fmaf(rv, sWT[k * WTS + 256 + f], fw2);
    }
    fw0 *= c; fw1 *= c; fw2 *= c;

    const size_t sb = (size_t)si * 384;
    const float g_sv = fw0 * so[sb + f];
    const float g_ev = fw1 * so[sb + 128 + f];
    const float g_ns = fw2 * so[sb + 256 + f];

    atomAddF(sacc + (size_t)ri * FDIM + f, s[(size_t)si * FDIM + f] * g_ns);
#pragma unroll
    for (int d = 0; d < 3; ++d) {
      const float mv = v[((size_t)si * 3 + d) * FDIM + f] * g_sv
                     + g_ev * sDir[el * 3 + d];
      atomAddF(vacc + ((size_t)ri * 3 + d) * FDIM + f, mv);
    }
  }
}

// ====================================================================
// Vn = ||Vv|| over d;  m_bf = [ s_acc | Vn ]  (N x 256 bf16)
// ====================================================================
__global__ void painn_vn_m(const float* __restrict__ sacc,
                           const float* __restrict__ Vv,
                           unsigned short* __restrict__ mbf)
{
  int t = blockIdx.x * blockDim.x + threadIdx.x;
  if (t >= NTOT * FDIM) return;
  const int n = t >> 7, f = t & 127;
  const size_t b3 = (size_t)n * 384 + f;
  const float v0 = Vv[b3], v1 = Vv[b3 + 128], v2 = Vv[b3 + 256];
  mbf[(size_t)n * 256 + f]       = f2bf(sacc[t]);
  mbf[(size_t)n * 256 + 128 + f] = f2bf(sqrtf(v0*v0 + v1*v1 + v2*v2));
}

// ====================================================================
// Final update: gated residual; writes next-layer s (f32+bf16), v (f32)
// and the output slice out[l, n, 0..3, f].
// ====================================================================
__global__ void painn_update(const float* __restrict__ sacc,
                             const float* __restrict__ vacc,
                             const float* __restrict__ Uv,
                             const float* __restrict__ Vv,
                             const float* __restrict__ m2,
                             float* __restrict__ s,
                             unsigned short* __restrict__ sbf,
                             float* __restrict__ v,
                             float* __restrict__ out, int l)
{
  int t = blockIdx.x * blockDim.x + threadIdx.x;
  if (t >= NTOT * FDIM) return;
  const int n = t >> 7, f = t & 127;
  const size_t b3 = (size_t)n * 384 + f;
  const float a_ss = m2[b3], a_sv = m2[b3 + 128], a_vv = m2[b3 + 256];
  float dot = Uv[b3] * Vv[b3] + Uv[b3 + 128] * Vv[b3 + 128]
            + Uv[b3 + 256] * Vv[b3 + 256];
  const float sn = sacc[t] + a_ss + a_sv * dot;
  s[t] = sn; sbf[t] = f2bf(sn);
  const size_t ob = ((size_t)l * NTOT + n) * 512 + f;
  out[ob] = sn;
#pragma unroll
  for (int d = 0; d < 3; ++d) {
    const size_t vi = b3 + (size_t)d * 128;
    const float nv = vacc[vi] + a_vv * Uv[vi];
    v[vi] = nv;
    out[ob + (size_t)(1 + d) * 128] = nv;
  }
}

// ====================================================================
// Host orchestration
// ====================================================================
extern "C" void kernel_launch(void* const* d_in, const int* in_sizes, int n_in,
                              void* d_out, int out_size, void* d_ws, size_t ws_size,
                              hipStream_t stream)
{
  (void)in_sizes; (void)n_in; (void)out_size; (void)ws_size;
  const float* atom_xyz = (const float*)d_in[0];
  const float* cell     = (const float*)d_in[1];
  const int*   nodes    = (const int*)  d_in[2];
  const int*   aedges   = (const int*)  d_in[3];
  const float* edisp    = (const float*)d_in[4];
  const float* Vflat    = (const float*)d_in[5];
  const float* Eflat    = (const float*)d_in[6];
  const float* atom_emb = (const float*)d_in[7];
  const float* pot_W    = (const float*)d_in[8];
  const float* pot_b    = (const float*)d_in[9];
  const float* init_W1  = (const float*)d_in[10];
  const float* init_b1  = (const float*)d_in[11];
  const float* init_W2  = (const float*)d_in[12];
  const float* init_b2  = (const float*)d_in[13];
  const float* vec_w    = (const float*)d_in[14];
  const float* filt_W   = (const float*)d_in[15];
  const float* filt_b   = (const float*)d_in[16];
  const float* msg_W1   = (const float*)d_in[17];
  const float* msg_b1   = (const float*)d_in[18];
  const float* msg_W2   = (const float*)d_in[19];
  const float* msg_b2   = (const float*)d_in[20];
  const float* U_W      = (const float*)d_in[21];
  const float* V_W      = (const float*)d_in[22];
  const float* upd_W1   = (const float*)d_in[23];
  const float* upd_b1   = (const float*)d_in[24];
  const float* upd_W2   = (const float*)d_in[25];
  const float* upd_b2   = (const float*)d_in[26];
  float* out = (float*)d_out;

  const size_t NF  = (size_t)NTOT * FDIM;   // 2,097,152
  const size_t NF3 = 3 * NF;

  char* wsb = (char*)d_ws;
  size_t off = 0;
  auto alloc = [&](size_t bytes) -> char* {
    off = (off + 255) & ~(size_t)255;
    char* p = wsb + off; off += bytes; return p;
  };
  float* s    = (float*)alloc(NF * 4);
  unsigned short* sbf    = (unsigned short*)alloc(NF * 2);
  float* v    = (float*)alloc(NF3 * 4);
  float* sacc = (float*)alloc(NF * 4);
  float* vacc = (float*)alloc(NF3 * 4);
  unsigned short* vaccbf = (unsigned short*)alloc(NF3 * 2);
  float* so   = (float*)alloc(NF3 * 4);
  unsigned short* tbf    = (unsigned short*)alloc(NF * 2);
  unsigned short* mbf    = (unsigned short*)alloc(2 * NF * 2);  // also init h
  float* m2   = (float*)alloc(NF3 * 4);
  float* Uv   = (float*)alloc(NF3 * 4);
  float* Vv   = (float*)alloc(NF3 * 4);
  float* rbf  = (float*)alloc((size_t)ETOT * KRBF * 4);
  float* cutv = (float*)alloc((size_t)ETOT * 4);
  float* dirn = (float*)alloc((size_t)ETOT * 3 * 4);
  int*   idxS = (int*)alloc((size_t)ETOT * 4);
  int*   idxR = (int*)alloc((size_t)ETOT * 4);
  unsigned short* iW1b = (unsigned short*)alloc((size_t)128*256*2);
  unsigned short* iW2b = (unsigned short*)alloc((size_t)128*128*2);
  unsigned short* mW1b = (unsigned short*)alloc((size_t)3*128*128*2);
  unsigned short* mW2b = (unsigned short*)alloc((size_t)3*384*128*2);
  unsigned short* uWb  = (unsigned short*)alloc((size_t)3*128*128*2);
  unsigned short* vWb  = (unsigned short*)alloc((size_t)3*128*128*2);
  unsigned short* pW1b = (unsigned short*)alloc((size_t)3*128*256*2);
  unsigned short* pW2b = (unsigned short*)alloc((size_t)3*384*128*2);

  auto cvt = [&](const float* src, unsigned short* dst, size_t n) {
    painn_cvt_bf16<<<(unsigned)((n + 255) / 256), 256, 0, stream>>>(src, dst, (int)n);
  };
  auto gemm = [&](const unsigned short* A, const unsigned short* W,
                  const float* bias, float* oF, unsigned short* oB,
                  int M, int Nd, int Kd, int act) {
    const int blocks = (M / 256) * (Nd / 64);
    painn_gemm_bf16<<<blocks, 256, 0, stream>>>(A, W, bias, oF, oB,
                                                M, Nd, Kd, act);
  };

  // ---- weight conversion (cheap, deterministic each call) ----
  cvt(init_W1, iW1b, (size_t)128*256);
  cvt(init_W2, iW2b, (size_t)128*128);
  cvt(msg_W1,  mW1b, (size_t)3*128*128);
  cvt(msg_W2,  mW2b, (size_t)3*384*128);
  cvt(U_W,     uWb,  (size_t)3*128*128);
  cvt(V_W,     vWb,  (size_t)3*128*128);
  cvt(upd_W1,  pW1b, (size_t)3*128*256);
  cvt(upd_W2,  pW2b, (size_t)3*384*128);

  // ---- edge geometry ----
  painn_edge_geom<<<ETOT / 256, 256, 0, stream>>>(atom_xyz, cell, aedges, edisp,
                                                  rbf, cutv, dirn, idxS, idxR);

  // ---- node init: h -> silu GEMM -> GEMM -> s ; v init ----
  painn_h_build<<<(unsigned)(NF / 256), 256, 0, stream>>>(nodes, atom_emb, Vflat,
                                                          pot_W, pot_b, mbf);
  gemm(mbf, iW1b, init_b1, nullptr, tbf, NTOT, 128, 256, 1);
  gemm(tbf, iW2b, init_b2, s, sbf, NTOT, 128, 128, 0);
  painn_v_init<<<(unsigned)(NF3 / 256), 256, 0, stream>>>(Eflat, vec_w, v);

  // ---- message-passing layers ----
  for (int l = 0; l < NLAY; ++l) {
    // so = silu(s @ msg_W1^T + b1) @ msg_W2^T + b2
    gemm(sbf, mW1b + (size_t)l*128*128, msg_b1 + l*128, nullptr, tbf,
         NTOT, 128, 128, 1);
    gemm(tbf, mW2b + (size_t)l*384*128, msg_b2 + l*384, so, nullptr,
         NTOT, 384, 128, 0);

    // accumulators = current s, v
    hipMemcpyAsync(sacc, s, NF * 4,  hipMemcpyDeviceToDevice, stream);
    hipMemcpyAsync(vacc, v, NF3 * 4, hipMemcpyDeviceToDevice, stream);

    // fused filter GEMM + gating + scatter
    painn_edge_msg<<<ETOT / EPB, 256, 0, stream>>>(
        rbf, cutv, dirn, idxS, idxR, so, s, v,
        filt_W + (size_t)l*384*KRBF, filt_b + l*384, sacc, vacc);

    // Uv / Vv GEMMs over flattened (3N, F)
    cvt(vacc, vaccbf, NF3);
    gemm(vaccbf, uWb + (size_t)l*128*128, nullptr, Uv, nullptr,
         3*NTOT, 128, 128, 0);
    gemm(vaccbf, vWb + (size_t)l*128*128, nullptr, Vv, nullptr,
         3*NTOT, 128, 128, 0);

    // m = [s | ||Vv||]; update MLP
    painn_vn_m<<<(unsigned)(NF / 256), 256, 0, stream>>>(sacc, Vv, mbf);
    gemm(mbf, pW1b + (size_t)l*128*256, upd_b1 + l*128, nullptr, tbf,
         NTOT, 128, 256, 1);
    gemm(tbf, pW2b + (size_t)l*384*128, upd_b2 + l*384, m2, nullptr,
         NTOT, 384, 128, 0);

    // gated residual + emit output slice
    painn_update<<<(unsigned)(NF / 256), 256, 0, stream>>>(
        sacc, vacc, Uv, Vv, m2, s, sbf, v, out, l);
  }
}
